// CausalGroupedQueryPageAttention_57870389346731
// MI455X (gfx1250) — matrix operand order repaired and checked by reading, not compiled, not run on hardware
//
#include <hip/hip_runtime.h>
#include <hip/hip_bf16.h>
#include <cstdint>

#define N_QO_HEADS 32
#define N_KV_HEADS 8
#define HEAD_DIM   128
#define PAGE_SZ    16
#define NUM_BLK    512
#define BATCH      2
#define SEQ_LEN    2048
#define TOT        (BATCH * SEQ_LEN)
#define NB_PER_SEQ (SEQ_LEN / PAGE_SZ)      /* 128 blocks per sequence      */
#define QT2_PER_SEQ (SEQ_LEN / 32)          /* 64 double-q-tiles / sequence */
#define QSTRIDE    (N_QO_HEADS * HEAD_DIM)  /* 4096 */
#define KVSTRIDE   (N_KV_HEADS * HEAD_DIM)  /* 1024 */

typedef __attribute__((ext_vector_type(16))) __bf16 v16bf;
typedef __attribute__((ext_vector_type(8)))  float  v8f;

union Frag { v16bf v; uint4 q[2]; unsigned short u[16]; };

template <bool B> struct BoolC { static constexpr bool value = B; };

// fp32 -> bf16 via hardware conversion (RNE)
__device__ __forceinline__ unsigned short f2bf(float a) {
  return __builtin_bit_cast(unsigned short, (__bf16)a);
}
__device__ __forceinline__ unsigned pack2bf(float a, float b) {
  unsigned short l = __builtin_bit_cast(unsigned short, (__bf16)a);
  unsigned short h = __builtin_bit_cast(unsigned short, (__bf16)b);
  return (unsigned)l | ((unsigned)h << 16);
}

__global__ void inv_init_kernel(int* inv) {
  int i = blockIdx.x * blockDim.x + threadIdx.x;
  if (i < NUM_BLK * PAGE_SZ) inv[i] = -1;
}
__global__ void inv_scatter_kernel(int* inv, const int* __restrict__ slots) {
  int t = blockIdx.x * blockDim.x + threadIdx.x;
  if (t < TOT) inv[slots[t]] = t;
}

__global__ __launch_bounds__(256)
void paged_attn_kernel(const float* __restrict__ query,
                       const float* __restrict__ knew,
                       const float* __restrict__ vnew,
                       const float* __restrict__ kcache,
                       const float* __restrict__ vcache,
                       const int*   __restrict__ btab,
                       const int*   __restrict__ inv,
                       float*       __restrict__ out)
{
  // K tile row-major (K^T B-fragments = two 16B chunks/lane, 32B apart),
  // V tile transposed (V B-fragments = two 16B chunks/lane, 32B apart),
  // per-wave P buffer for C-layout -> A-layout conversion.
  __shared__ __align__(16) unsigned short Klds[32][HEAD_DIM];   // 8 KB
  __shared__ __align__(16) unsigned short Vtlds[HEAD_DIM][32];  // 8 KB
  __shared__ __align__(16) unsigned short Plds[8][16][32];      // 8 KB

  const int qt2 =  blockIdx.x       & (QT2_PER_SEQ - 1);  // 32-row double tile
  const int hkv = (blockIdx.x >> 6) & (N_KV_HEADS - 1);
  const int b   =  blockIdx.x >> 9;

  const int tid  = threadIdx.x;
  const int lane = tid & 31;
  const int wid  = tid >> 5;        // 0..7
  const int half = lane >> 4;       // lane half (0/1)
  const int ln   = lane & 15;

  const int hq   = hkv * 4 + (wid & 3);      // q-head within group
  const int qsub = wid >> 2;                 // which 16-row sub-tile
  const int sq0  = qt2 * 32 + qsub * 16;     // first q position (in-sequence)

  // scores in log2 domain: fold 1/sqrt(d)*log2(e) into Q before bf16 rounding
  const float SCL2 = 0.08838834764831845f * 1.4426950408889634f;

  // ---- preload Q (pre-scaled) as 4 bf16 A-fragments (16x32, K = d chunk) ----
  Frag qa[4];
  {
    const float* qrow = query + (size_t)(b * SEQ_LEN + sq0 + ln) * QSTRIDE + hq * HEAD_DIM;
    #pragma unroll
    for (int c = 0; c < 4; ++c) {
      const int d0 = c * 32 + half * 8;        // A chunk0: K = d0..d0+7
      const int d1 = c * 32 + 16 + half * 8;   // A chunk1: K = d1..d1+7
      #pragma unroll
      for (int i = 0; i < 8; ++i) {
        qa[c].u[i]     = f2bf(qrow[d0 + i] * SCL2);
        qa[c].u[8 + i] = f2bf(qrow[d1 + i] * SCL2);
      }
    }
  }

  // constant all-ones bf16 B fragment: row-sum(P) = P x ones via one WMMA
  Frag onesb;
  #pragma unroll
  for (int i = 0; i < 16; ++i) onesb.u[i] = 0x3F80;

  // ---- flash-attention state (16 q-rows x 128 d, fp32) ----
  v8f o[8];
  float m_r[8], l_r[8];
  #pragma unroll
  for (int t = 0; t < 8; ++t) { v8f z = {}; o[t] = z; }
  #pragma unroll
  for (int r = 0; r < 8; ++r) { m_r[r] = -__builtin_inff(); l_r[r] = 0.0f; }

  // cooperative staging: 8 threads per key, 16 d-values each
  const int kk = tid >> 3;
  const int dq = (tid & 7) * 16;

  // one key-tile step; DIAG (causal masking) specialized at compile time
  auto step = [&](int kt, auto diag_c) {
    constexpr bool DIAG = decltype(diag_c)::value;

    __syncthreads();   // previous iteration done reading K/V tiles

    // ---- gather K/V tile (paged cache with new-token override) -> bf16 LDS --
    {
      const int s    = kt * 32 + kk;                       // key pos in sequence
      const int blk  = btab[b * NB_PER_SEQ + (s >> 4)];
      const int slot = blk * PAGE_SZ + (s & (PAGE_SZ - 1));
      const int tnew = inv[slot];
      const float* ksrc = (tnew >= 0) ? (knew   + (size_t)tnew * KVSTRIDE + hkv * HEAD_DIM)
                                      : (kcache + (size_t)slot * KVSTRIDE + hkv * HEAD_DIM);
      const float* vsrc = (tnew >= 0) ? (vnew   + (size_t)tnew * KVSTRIDE + hkv * HEAD_DIM)
                                      : (vcache + (size_t)slot * KVSTRIDE + hkv * HEAD_DIM);
      const float4* k4 = reinterpret_cast<const float4*>(ksrc + dq);
      const float4* v4 = reinterpret_cast<const float4*>(vsrc + dq);
      #pragma unroll
      for (int i = 0; i < 4; ++i) {
        float4 kv = k4[i], vv = v4[i];
        const int d = dq + i * 4;
        *(unsigned*)&Klds[kk][d]     = pack2bf(kv.x, kv.y);
        *(unsigned*)&Klds[kk][d + 2] = pack2bf(kv.z, kv.w);
        const unsigned v01 = pack2bf(vv.x, vv.y);
        const unsigned v23 = pack2bf(vv.z, vv.w);
        Vtlds[d + 0][kk] = (unsigned short)v01;
        Vtlds[d + 1][kk] = (unsigned short)(v01 >> 16);
        Vtlds[d + 2][kk] = (unsigned short)v23;
        Vtlds[d + 3][kk] = (unsigned short)(v23 >> 16);
      }
    }
    __syncthreads();

    // ---- S = Q * K^T : 16x32 tile = two 16x16 WMMA C tiles (log2 domain) ----
    v8f s0 = {}, s1 = {};
    #pragma unroll
    for (int c = 0; c < 4; ++c) {
      Frag kb0, kb1;
      const uint4* k0p = reinterpret_cast<const uint4*>(&Klds[ln][c * 32 + half * 8]);
      const uint4* k1p = reinterpret_cast<const uint4*>(&Klds[16 + ln][c * 32 + half * 8]);
      kb0.q[0] = k0p[0]; kb0.q[1] = k0p[2];
      kb1.q[0] = k1p[0]; kb1.q[1] = k1p[2];
      s0 = __builtin_amdgcn_wmma_f32_16x16x32_bf16(false, qa[c].v, false, kb0.v, (short)0, s0, false, false);
      s1 = __builtin_amdgcn_wmma_f32_16x16x32_bf16(false, qa[c].v, false, kb1.v, (short)0, s1, false, false);
    }

    // ---- online softmax over the 32 keys of this tile ----
    float sc_r[8];
    #pragma unroll
    for (int r = 0; r < 8; ++r) {
      float a0 = s0[r];
      float a1 = s1[r];
      if constexpr (DIAG) {
        const int qpos = sq0 + r + 8 * half;
        if (kt * 32 + ln > qpos)      a0 = -__builtin_inff();
        if (kt * 32 + 16 + ln > qpos) a1 = -__builtin_inff();
      }
      float mx = fmaxf(a0, a1);
      mx = fmaxf(mx, __shfl_xor(mx, 1));
      mx = fmaxf(mx, __shfl_xor(mx, 2));
      mx = fmaxf(mx, __shfl_xor(mx, 4));
      mx = fmaxf(mx, __shfl_xor(mx, 8));
      const float mnew = fmaxf(m_r[r], mx);
      const float sc = exp2f(m_r[r] - mnew);
      const float p0 = exp2f(a0 - mnew);
      const float p1 = exp2f(a1 - mnew);
      sc_r[r] = sc;
      m_r[r] = mnew;
      // C layout: (row = r + 8*half, col = ln / 16+ln) -> P buffer row-major
      Plds[wid][r + 8 * half][ln]      = f2bf(p0);
      Plds[wid][r + 8 * half][16 + ln] = f2bf(p1);
      #pragma unroll
      for (int t = 0; t < 8; ++t) o[t][r] *= sc;
    }

    // P stored via cross-lane C layout; fence LDS before A-layout reload.
    __asm__ volatile("s_wait_dscnt 0x0" ::: "memory");

    // ---- A fragment of P: lane -> row ln, chunks (K = half*8, 16+half*8) ----
    Frag pf;
    {
      const uint4* p0p = reinterpret_cast<const uint4*>(&Plds[wid][ln][half * 8]);
      const uint4* p1p = reinterpret_cast<const uint4*>(&Plds[wid][ln][16 + half * 8]);
      pf.q[0] = p0p[0];
      pf.q[1] = p1p[0];
    }

    // ---- row-sum(P) via matrix core: one WMMA against all-ones B ----
    v8f zsum = {};
    zsum = __builtin_amdgcn_wmma_f32_16x16x32_bf16(false, pf.v, false, onesb.v, (short)0, zsum, false, false);

    // ---- O += P * V : 8 d-tiles of 16 ----
    #pragma unroll
    for (int t = 0; t < 8; ++t) {
      Frag vb;
      const uint4* vp = reinterpret_cast<const uint4*>(&Vtlds[t * 16 + ln][half * 8]);
      vb.q[0] = vp[0]; vb.q[1] = vp[2];
      o[t] = __builtin_amdgcn_wmma_f32_16x16x32_bf16(false, pf.v, false, vb.v, (short)0, o[t], false, false);
    }

    // ---- l update with WMMA row-sums (same bf16 P as the numerator) ----
    #pragma unroll
    for (int r = 0; r < 8; ++r) l_r[r] = l_r[r] * sc_r[r] + zsum[r];
  };

  // steady-state tiles: no causal masking; diagonal tile peeled (masked).
  // Trip count (qt2+1) is uniform across all 8 waves -> barriers are safe.
  for (int kt = 0; kt < qt2; ++kt) step(kt, BoolC<false>{});
  step(qt2, BoolC<true>{});

  // ---- epilogue: O / l, write fp32 output ----
  #pragma unroll
  for (int r = 0; r < 8; ++r) {
    const float ilv = 1.0f / l_r[r];
    const size_t rowbase = (size_t)(b * SEQ_LEN + sq0 + r + 8 * half) * QSTRIDE + hq * HEAD_DIM;
    #pragma unroll
    for (int t = 0; t < 8; ++t) {
      out[rowbase + t * 16 + ln] = o[t][r] * ilv;
    }
  }
}

extern "C" void kernel_launch(void* const* d_in, const int* in_sizes, int n_in,
                              void* d_out, int out_size, void* d_ws, size_t ws_size,
                              hipStream_t stream) {
  const float* query  = (const float*)d_in[0];
  const float* knew   = (const float*)d_in[1];
  const float* vnew   = (const float*)d_in[2];
  const float* kcache = (const float*)d_in[3];
  const float* vcache = (const float*)d_in[4];
  const int*   btab   = (const int*)d_in[5];
  const int*   slots  = (const int*)d_in[6];
  float* out = (float*)d_out;

  int* inv = (int*)d_ws;   // NUM_BLK*PAGE_SZ ints = 32 KB slot->token inverse map

  inv_init_kernel<<<(NUM_BLK * PAGE_SZ + 255) / 256, 256, 0, stream>>>(inv);
  inv_scatter_kernel<<<(TOT + 255) / 256, 256, 0, stream>>>(inv, slots);

  dim3 grid(BATCH * N_KV_HEADS * QT2_PER_SEQ);   // 1024 workgroups, 8 waves each
  paged_attn_kernel<<<grid, 256, 0, stream>>>(query, knew, vnew, kcache, vcache,
                                              btab, inv, out);
}
